// MusicVAEDecoder_53412213293364
// MI455X (gfx1250) — matrix operand order repaired and checked by reading, not compiled, not run on hardware
//
#include <hip/hip_runtime.h>
#include <cstdint>

// MusicVAE 3-layer LSTM decoder for MI455X (gfx1250).
// bf16 weights repacked once into WMMA B-fragment layout (~178 MB -> L2-resident
// across all 32 steps), per-step GEMMs on v_wmma_f32_16x16x32_bf16 with fp32
// accumulation. A-tiles (activations) staged into LDS with gfx1250 async
// global->LDS copies (ASYNCcnt), double-buffered, shared by all 8 waves/block.

typedef __attribute__((ext_vector_type(16))) __bf16 v16bf;
typedef __attribute__((ext_vector_type(8)))  float  v8f;

#define B_   64
#define T_   32
#define D_   90
#define Z_   512
#define U_   2048
#define NG_  8192      // 4*U
#define KX_  608       // D+Z (602) padded to multiple of 32
#define KTX_ 19        // KX_/32
#define KTU_ 64        // U_/32
#define NTG_ 512       // NG_/16
#define NTD_ 6         // 96/16 (D padded 90->96)

#define WMMA_BF16(a, b, c) \
  __builtin_amdgcn_wmma_f32_16x16x32_bf16(false, (a), false, (b), (short)0, (c), false, false)

#if __has_builtin(__builtin_amdgcn_global_load_async_to_lds_b128)
#define ASYNC_LDS 1
#else
#define ASYNC_LDS 0
#endif

// b128 async copy operates on 16-byte int vectors (per hipcc diagnostic).
typedef int v4i_ __attribute__((vector_size(16)));
typedef __attribute__((address_space(1))) v4i_* gas1_v4i;
typedef __attribute__((address_space(3))) v4i_* las3_v4i;

static __device__ __forceinline__ unsigned short f2bf(float f) {
  union { float f; unsigned u; } v; v.f = f;
  unsigned u = v.u;
  u += 0x7FFFu + ((u >> 16) & 1u);   // round-to-nearest-even
  return (unsigned short)(u >> 16);
}

// Index into packed A-fragment layout [(K/32)][Mtile=4][lane=32][16 bf16]
// ISA 16-bit A 16x32: lanes 0-15 hold K%32 in {0..7,16..23}, lanes 16-31 in {8..15,24..31}.
static __device__ __forceinline__ size_t apack_idx(int kt, int M, int ko) {
  int mt = M >> 4, m = M & 15;
  int lane = m + (((ko >> 3) & 1) << 4);
  int pos  = (ko & 7) + ((ko >> 4) << 3);
  return ((size_t)(kt * 4 + mt) * 32 + lane) * 16 + pos;
}

// ---- cooperative stage of one 4KB A-k-tile into LDS (async path on gfx1250) ----
static __device__ __forceinline__ void stage_a(const unsigned short* __restrict__ A, int kt,
                                               unsigned short* sdst, int tid) {
  const unsigned short* src = A + (size_t)kt * 2048 + (size_t)tid * 8;  // 16B per lane
  unsigned short* dst = sdst + (size_t)tid * 8;
#if ASYNC_LDS
  __builtin_amdgcn_global_load_async_to_lds_b128(
      (gas1_v4i)(uintptr_t)src, (las3_v4i)(unsigned)(uintptr_t)dst, 0, 0);
#else
  *(uint4*)dst = *(const uint4*)src;
#endif
}

static __device__ __forceinline__ void wait_stage() {
#if ASYNC_LDS
#if __has_builtin(__builtin_amdgcn_s_wait_asynccnt)
  __builtin_amdgcn_s_wait_asynccnt(0);
#else
  asm volatile("s_wait_asynccnt 0x0" ::: "memory");
#endif
#endif
  __syncthreads();
}

// One GEMM term: acc += A(packed, KT k-tiles) @ W(packed B-fragments, NTG_ strips)
static __device__ __forceinline__ void run_term(const unsigned short* __restrict__ A, int KT,
                                                const unsigned short* __restrict__ Wp,
                                                int nt, int lane, int tid,
                                                unsigned short (&sA)[2][2048],
                                                v8f (&acc)[4]) {
  stage_a(A, 0, sA[0], tid);
  const unsigned short* bbase = Wp + (size_t)nt * 512 + (size_t)lane * 16;
  v16bf bcur = *(const v16bf*)bbase;                     // k-tile 0 weight fragment
  for (int kt = 0; kt < KT; ++kt) {
    const int buf = kt & 1;
    wait_stage();                                        // sA[buf] ready for all waves
    if (kt + 1 < KT) stage_a(A, kt + 1, sA[buf ^ 1], tid);  // overlap next stage
    v16bf bnext = bcur;
    if (kt + 1 < KT) {                                   // one-tile register lookahead on B
      const unsigned short* bp = bbase + (size_t)(kt + 1) * 262144;
      __builtin_prefetch((const void*)(bp + 262144), 0, 1);
      bnext = *(const v16bf*)bp;
    }
    const unsigned short* sp = &sA[buf][lane * 16];      // ds_load_b128 x2 per fragment
    v16bf a0 = *(const v16bf*)(sp);
    v16bf a1 = *(const v16bf*)(sp + 512);
    v16bf a2 = *(const v16bf*)(sp + 1024);
    v16bf a3 = *(const v16bf*)(sp + 1536);
    acc[0] = WMMA_BF16(a0, bcur, acc[0]);
    acc[1] = WMMA_BF16(a1, bcur, acc[1]);
    acc[2] = WMMA_BF16(a2, bcur, acc[2]);
    acc[3] = WMMA_BF16(a3, bcur, acc[3]);
    bcur = bnext;
  }
  __syncthreads();   // everyone done reading sA before next term restages it
}

// ---------------- one-time repack: fp32 W[K,N] -> bf16 B-fragments ----------------
// Layout [(K/32)][(N/16)][lane=32][16 bf16]; N=nt*16+lane%16;
// K=kt*32+8*(lane/16)+(pos&7)+16*(pos>>3).  Zero-pad outside K,N.
__global__ void repack_w(const float* __restrict__ W, unsigned short* __restrict__ out,
                         int K, int N, int KT, int NT) {
  size_t id = (size_t)blockIdx.x * blockDim.x + threadIdx.x;
  size_t total = (size_t)KT * NT * 32;
  if (id >= total) return;
  int lane = (int)(id & 31);
  size_t frag = id >> 5;
  int nt = (int)(frag % NT);
  int kt = (int)(frag / NT);
  int n = nt * 16 + (lane & 15);
  int kbase = kt * 32 + ((lane >> 4) << 3);
  __attribute__((aligned(16))) unsigned short vals[16];
#pragma unroll
  for (int p = 0; p < 16; ++p) {
    int k = kbase + (p & 7) + ((p >> 3) << 4);
    float f = (k < K && n < N) ? W[(size_t)k * N + n] : 0.0f;
    vals[p] = f2bf(f);
  }
  uint4* d4 = (uint4*)(out + (frag * 32 + lane) * 16);
  d4[0] = ((const uint4*)vals)[0];
  d4[1] = ((const uint4*)vals)[1];
}

// ---------------- init = z @ Wz + bz -> (h,c) per layer (one-time, fp32) ----------
__global__ void init_state(const float* __restrict__ z, const float* __restrict__ Wz,
                           const float* __restrict__ bz,
                           unsigned short* __restrict__ Hp0, unsigned short* __restrict__ Hp1,
                           unsigned short* __restrict__ Hp2,
                           float* __restrict__ c0, float* __restrict__ c1, float* __restrict__ c2) {
  const int NC = 3 * 2 * U_;  // 12288
  int id = blockIdx.x * blockDim.x + threadIdx.x;
  if (id >= B_ * NC) return;
  int col = id % NC;
  int b   = id / NC;
  float acc = bz[col];
  const float* zr = z + (size_t)b * Z_;
  for (int k = 0; k < Z_; ++k) acc += zr[k] * Wz[(size_t)k * NC + col];
  int l  = col >> 12;          // /4096
  int hc = (col >> 11) & 1;
  int u  = col & (U_ - 1);
  if (hc) {
    float* c = (l == 0) ? c0 : (l == 1) ? c1 : c2;
    c[(size_t)b * U_ + u] = acc;
  } else {
    unsigned short* Hp = (l == 0) ? Hp0 : (l == 1) ? Hp1 : Hp2;
    Hp[apack_idx(u >> 5, b, u & 31)] = f2bf(acc);
  }
}

// ---------------- pack all T inputs x_t = [teacher_t | z | 0pad] ------------------
__global__ void pack_xs(const float* __restrict__ teacher, const float* __restrict__ z,
                        unsigned short* __restrict__ Xp) {
  int id = blockIdx.x * blockDim.x + threadIdx.x;
  if (id >= T_ * B_ * KX_) return;
  int k   = id % KX_;
  int rem = id / KX_;
  int b   = rem % B_;
  int t   = rem / B_;
  float v = 0.0f;
  if (k < D_)            v = teacher[((size_t)b * T_ + t) * D_ + k];
  else if (k < D_ + Z_)  v = z[(size_t)b * Z_ + (k - D_)];
  Xp[(size_t)t * (KTX_ * 2048) + apack_idx(k >> 5, b, k & 31)] = f2bf(v);
}

// ---------------- G[64,8192] = Ax @ Wp + Ah @ Up (WMMA bf16, LDS-staged A) --------
__global__ void __launch_bounds__(256) gemm_gates(
    const unsigned short* __restrict__ Ax, int ktx,
    const unsigned short* __restrict__ Wp,
    const unsigned short* __restrict__ Ah,
    const unsigned short* __restrict__ Up,
    float* __restrict__ G) {
  __shared__ __attribute__((aligned(32))) unsigned short sA[2][2048];  // 2 x 4KB
  const int tid  = threadIdx.x;
  const int lane = tid & 31;
  const int nt   = blockIdx.x * 8 + (tid >> 5);   // 0..511
  v8f zv = {};
  v8f acc[4] = {zv, zv, zv, zv};

  run_term(Ax, ktx,  Wp, nt, lane, tid, sA, acc);   // W-term
  run_term(Ah, KTU_, Up, nt, lane, tid, sA, acc);   // U-term

  // C/D layout: VGPR r, lane l -> row = r + (l<16?0:8), col = l%16
  const int col  = nt * 16 + (lane & 15);
  const int rofs = (lane >> 4) << 3;
#pragma unroll
  for (int m = 0; m < 4; ++m) {
#pragma unroll
    for (int r = 0; r < 8; ++r) {
      G[(size_t)(m * 16 + r + rofs) * NG_ + col] = acc[m][r];
    }
  }
}

// ---------------- LSTM gate nonlinearities + state update -------------------------
__global__ void gates_kernel(const float* __restrict__ G, const float* __restrict__ bias,
                             float* __restrict__ c, unsigned short* __restrict__ Hp,
                             unsigned short* __restrict__ Hp2) {
  int id = blockIdx.x * blockDim.x + threadIdx.x;
  if (id >= B_ * U_) return;
  int u = id & (U_ - 1);
  int b = id >> 11;
  const float* g = G + (size_t)b * NG_;
  float gi = g[u]           + bias[u];
  float gf = g[U_ + u]      + bias[U_ + u];
  float gc = g[2 * U_ + u]  + bias[2 * U_ + u];
  float go = g[3 * U_ + u]  + bias[3 * U_ + u];
  float si = 1.0f / (1.0f + __expf(-gi));
  float sf = 1.0f / (1.0f + __expf(-gf));
  float so = 1.0f / (1.0f + __expf(-go));
  float c2 = sf * c[id] + si * tanhf(gc);
  float h2 = so * tanhf(c2);
  c[id] = c2;
  unsigned short hb = f2bf(h2);
  size_t pi = apack_idx(u >> 5, b, u & 31);
  Hp[pi] = hb;
  if (Hp2) Hp2[pi] = hb;    // layer-2: also keep per-t copy for the output GEMM
}

// ---------------- out[b,t,:] = h2_t @ Wout + bout (WMMA bf16) ---------------------
__global__ void __launch_bounds__(256) final_out(
    const unsigned short* __restrict__ H2all,
    const unsigned short* __restrict__ WoP,
    const float* __restrict__ bout,
    float* __restrict__ out) {
  const int lane = threadIdx.x & 31;
  const int gw   = blockIdx.x * 8 + (threadIdx.x >> 5);   // 0..T_*NTD_-1
  const int nt   = gw % NTD_;
  const int t    = gw / NTD_;
  v8f zv = {};
  v8f acc[4] = {zv, zv, zv, zv};
  const unsigned short* A = H2all + (size_t)t * (KTU_ * 2048);
  for (int kt = 0; kt < KTU_; ++kt) {
    const unsigned short* bp = WoP + (size_t)kt * (NTD_ * 512) + (size_t)nt * 512 + lane * 16;
    v16bf bf = *(const v16bf*)bp;
    const unsigned short* ap = A + (size_t)kt * 2048 + lane * 16;
    v16bf a0 = *(const v16bf*)(ap);
    v16bf a1 = *(const v16bf*)(ap + 512);
    v16bf a2 = *(const v16bf*)(ap + 1024);
    v16bf a3 = *(const v16bf*)(ap + 1536);
    acc[0] = WMMA_BF16(a0, bf, acc[0]);
    acc[1] = WMMA_BF16(a1, bf, acc[1]);
    acc[2] = WMMA_BF16(a2, bf, acc[2]);
    acc[3] = WMMA_BF16(a3, bf, acc[3]);
  }
  const int n = nt * 16 + (lane & 15);
  if (n < D_) {
    const int rofs = (lane >> 4) << 3;
    float bo = bout[n];
#pragma unroll
    for (int m = 0; m < 4; ++m) {
#pragma unroll
      for (int r = 0; r < 8; ++r) {
        int b = m * 16 + r + rofs;
        out[((size_t)b * T_ + t) * D_ + n] = acc[m][r] + bo;
      }
    }
  }
}

extern "C" void kernel_launch(void* const* d_in, const int* in_sizes, int n_in,
                              void* d_out, int out_size, void* d_ws, size_t ws_size,
                              hipStream_t stream) {
  (void)in_sizes; (void)n_in; (void)out_size; (void)ws_size;
  const float* z       = (const float*)d_in[0];
  const float* teacher = (const float*)d_in[1];
  const float* Wz      = (const float*)d_in[2];
  const float* bz      = (const float*)d_in[3];
  const float* W0      = (const float*)d_in[4];
  const float* U0w     = (const float*)d_in[5];
  const float* b0      = (const float*)d_in[6];
  const float* W1      = (const float*)d_in[7];
  const float* U1w     = (const float*)d_in[8];
  const float* b1      = (const float*)d_in[9];
  const float* W2      = (const float*)d_in[10];
  const float* U2w     = (const float*)d_in[11];
  const float* b2      = (const float*)d_in[12];
  const float* Wout    = (const float*)d_in[13];
  const float* bout    = (const float*)d_in[14];
  float* out = (float*)d_out;

  char* ws = (char*)d_ws;
  size_t off = 0;
  auto alloc = [&](size_t bytes) -> void* {
    void* p = ws + off;
    off += (bytes + 255) & ~(size_t)255;
    return p;
  };
  // bf16 weight fragments (~178 MB total -> L2-resident across all 32 steps)
  unsigned short* Wp0 = (unsigned short*)alloc((size_t)KTX_ * NTG_ * 512 * 2);
  unsigned short* Up0 = (unsigned short*)alloc((size_t)KTU_ * NTG_ * 512 * 2);
  unsigned short* Wp1 = (unsigned short*)alloc((size_t)KTU_ * NTG_ * 512 * 2);
  unsigned short* Up1 = (unsigned short*)alloc((size_t)KTU_ * NTG_ * 512 * 2);
  unsigned short* Wp2 = (unsigned short*)alloc((size_t)KTU_ * NTG_ * 512 * 2);
  unsigned short* Up2 = (unsigned short*)alloc((size_t)KTU_ * NTG_ * 512 * 2);
  unsigned short* WoP = (unsigned short*)alloc((size_t)KTU_ * NTD_ * 512 * 2);
  unsigned short* Xp  = (unsigned short*)alloc((size_t)T_ * KTX_ * 2048 * 2);
  unsigned short* Hp0 = (unsigned short*)alloc((size_t)KTU_ * 2048 * 2);
  unsigned short* Hp1 = (unsigned short*)alloc((size_t)KTU_ * 2048 * 2);
  unsigned short* Hp2 = (unsigned short*)alloc((size_t)KTU_ * 2048 * 2);
  unsigned short* H2a = (unsigned short*)alloc((size_t)T_ * KTU_ * 2048 * 2);
  float* c0 = (float*)alloc((size_t)B_ * U_ * 4);
  float* c1 = (float*)alloc((size_t)B_ * U_ * 4);
  float* c2 = (float*)alloc((size_t)B_ * U_ * 4);
  float* G  = (float*)alloc((size_t)B_ * NG_ * 4);

  const int TB = 256;
  auto grid = [](size_t n, int tb) { return (unsigned)((n + (size_t)tb - 1) / tb); };

  // one-time weight repack (fp32 -> bf16 B-fragments)
  repack_w<<<grid((size_t)KTX_ * NTG_ * 32, TB), TB, 0, stream>>>(W0,  Wp0, D_ + Z_, NG_, KTX_, NTG_);
  repack_w<<<grid((size_t)KTU_ * NTG_ * 32, TB), TB, 0, stream>>>(U0w, Up0, U_, NG_, KTU_, NTG_);
  repack_w<<<grid((size_t)KTU_ * NTG_ * 32, TB), TB, 0, stream>>>(W1,  Wp1, U_, NG_, KTU_, NTG_);
  repack_w<<<grid((size_t)KTU_ * NTG_ * 32, TB), TB, 0, stream>>>(U1w, Up1, U_, NG_, KTU_, NTG_);
  repack_w<<<grid((size_t)KTU_ * NTG_ * 32, TB), TB, 0, stream>>>(W2,  Wp2, U_, NG_, KTU_, NTG_);
  repack_w<<<grid((size_t)KTU_ * NTG_ * 32, TB), TB, 0, stream>>>(U2w, Up2, U_, NG_, KTU_, NTG_);
  repack_w<<<grid((size_t)KTU_ * NTD_ * 32, TB), TB, 0, stream>>>(Wout, WoP, U_, D_, KTU_, NTD_);

  // initial state + all teacher-forcing inputs
  init_state<<<grid((size_t)B_ * 12288, TB), TB, 0, stream>>>(z, Wz, bz, Hp0, Hp1, Hp2, c0, c1, c2);
  pack_xs<<<grid((size_t)T_ * B_ * KX_, TB), TB, 0, stream>>>(teacher, z, Xp);

  // sequential time loop: stream order provides the t / layer dependencies
  for (int t = 0; t < T_; ++t) {
    gemm_gates<<<NTG_ / 8, TB, 0, stream>>>(Xp + (size_t)t * KTX_ * 2048, KTX_, Wp0, Hp0, Up0, G);
    gates_kernel<<<grid((size_t)B_ * U_, TB), TB, 0, stream>>>(G, b0, c0, Hp0, (unsigned short*)nullptr);
    gemm_gates<<<NTG_ / 8, TB, 0, stream>>>(Hp0, KTU_, Wp1, Hp1, Up1, G);
    gates_kernel<<<grid((size_t)B_ * U_, TB), TB, 0, stream>>>(G, b1, c1, Hp1, (unsigned short*)nullptr);
    gemm_gates<<<NTG_ / 8, TB, 0, stream>>>(Hp1, KTU_, Wp2, Hp2, Up2, G);
    gates_kernel<<<grid((size_t)B_ * U_, TB), TB, 0, stream>>>(G, b2, c2, Hp2, H2a + (size_t)t * KTU_ * 2048);
  }

  // projection to output depth
  final_out<<<(T_ * NTD_) / 8, TB, 0, stream>>>(H2a, WoP, bout, out);
}
